// Conservative45KCNN_QNN_Model_22033182228575
// MI455X (gfx1250) — compile-verified, matrix operand order checked by reading, not run on hardware
//
#include <hip/hip_runtime.h>
#include <hip/hip_bf16.h>
#include <math.h>

typedef __attribute__((ext_vector_type(16))) _Float16 v16h;
typedef __attribute__((ext_vector_type(8)))  float    v8f;

// ---------------------------------------------------------------------------
// conv1: (4096,1,64,64) -> (4096,4,32,32), k=5, stride=2, pad=2, ReLU
// ---------------------------------------------------------------------------
__global__ __launch_bounds__(256) void conv1_relu(const float* __restrict__ x,
                                                  const float* __restrict__ w1,
                                                  const float* __restrict__ b1,
                                                  float* __restrict__ y1) {
    int o  = blockIdx.x * 256 + threadIdx.x;          // 4096*4*32*32 outputs
    int ox = o & 31, oy = (o >> 5) & 31, c = (o >> 10) & 3, b = o >> 12;
    const float* xin = x + (size_t)b * 4096;          // 64*64
    const float* w   = w1 + c * 25;
    float acc = b1[c];
    #pragma unroll
    for (int ky = 0; ky < 5; ++ky) {
        int iy = oy * 2 + ky - 2;
        if (iy < 0 || iy > 63) continue;
        #pragma unroll
        for (int kx = 0; kx < 5; ++kx) {
            int ix = ox * 2 + kx - 2;
            if (ix < 0 || ix > 63) continue;
            acc = fmaf(xin[iy * 64 + ix], w[ky * 5 + kx], acc);
        }
    }
    y1[o] = fmaxf(acc, 0.f);
}

// ---------------------------------------------------------------------------
// conv2: (4096,4,32,32) -> (4096,8,16,16), k=3, stride=2, pad=1, ReLU
// ---------------------------------------------------------------------------
__global__ __launch_bounds__(256) void conv2_relu(const float* __restrict__ y1,
                                                  const float* __restrict__ w2,
                                                  const float* __restrict__ b2,
                                                  float* __restrict__ y2) {
    int o  = blockIdx.x * 256 + threadIdx.x;          // 4096*8*16*16 outputs
    int ox = o & 15, oy = (o >> 4) & 15, c = (o >> 8) & 7, b = o >> 11;
    float acc = b2[c];
    #pragma unroll
    for (int ci = 0; ci < 4; ++ci) {
        const float* src = y1 + ((size_t)b * 4 + ci) * 1024;   // 32*32
        const float* w   = w2 + ((c * 4 + ci) * 9);
        #pragma unroll
        for (int ky = 0; ky < 3; ++ky) {
            int iy = oy * 2 + ky - 1;
            if (iy < 0 || iy > 31) continue;
            #pragma unroll
            for (int kx = 0; kx < 3; ++kx) {
                int ix = ox * 2 + kx - 1;
                if (ix < 0 || ix > 31) continue;
                acc = fmaf(src[iy * 32 + ix], w[ky * 3 + kx], acc);
            }
        }
    }
    y2[o] = fmaxf(acc, 0.f);
}

// ---------------------------------------------------------------------------
// conv3 (k=3,s=1,p=1, ReLU) fused with 8x8 mean-pool: one block per image.
// y2 image (8x16x16) + w3 staged in LDS; output h[b][64] (16ch x 2 x 2).
// ---------------------------------------------------------------------------
__global__ __launch_bounds__(256) void conv3_pool(const float* __restrict__ y2,
                                                  const float* __restrict__ w3,
                                                  const float* __restrict__ b3,
                                                  float* __restrict__ h) {
    __shared__ float img[8][16][16];     // 2048 floats
    __shared__ float wgt[16 * 8 * 9];    // 1152 floats
    __shared__ float part[256][2];
    int t = threadIdx.x, b = blockIdx.x;
    const float* src = y2 + (size_t)b * 2048;
    for (int i = t; i < 2048; i += 256) ((float*)img)[i] = src[i];
    for (int i = t; i < 1152; i += 256) wgt[i] = w3[i];
    __syncthreads();

    int c = t >> 4, row = t & 15;
    float bias = b3[c];
    float s0 = 0.f, s1 = 0.f;            // pool cols 0..7 / 8..15 of this row
    for (int px = 0; px < 16; ++px) {
        float acc = bias;
        for (int ci = 0; ci < 8; ++ci) {
            const float* wk = &wgt[(c * 8 + ci) * 9];
            #pragma unroll
            for (int ky = 0; ky < 3; ++ky) {
                int iy = row + ky - 1;
                if (iy < 0 || iy > 15) continue;
                #pragma unroll
                for (int kx = 0; kx < 3; ++kx) {
                    int ix = px + kx - 1;
                    if (ix < 0 || ix > 15) continue;
                    acc = fmaf(img[ci][iy][ix], wk[ky * 3 + kx], acc);
                }
            }
        }
        float r = fmaxf(acc, 0.f);       // ReLU before pooling (as reference)
        if (px < 8) s0 += r; else s1 += r;
    }
    part[t][0] = s0; part[t][1] = s1;
    __syncthreads();
    if (t < 64) {                        // (c, pool_row, pool_col)
        int cc = t >> 2, pr = (t >> 1) & 1, pc = t & 1;
        float acc = 0.f;
        for (int r = 0; r < 8; ++r) acc += part[cc * 16 + pr * 8 + r][pc];
        h[(size_t)b * 64 + cc * 4 + pr * 2 + pc] = acc * (1.f / 64.f);
    }
}

// ---------------------------------------------------------------------------
// FC: feats = relu(h @ wf^T + bf).  M=4096, N=256, K=64.
// One 16x16 tile per wave; K=64 as two v_wmma_f32_16x16x32_f16 steps.
// ---------------------------------------------------------------------------
__global__ __launch_bounds__(256) void fc_wmma(const float* __restrict__ h,
                                               const float* __restrict__ wf,
                                               const float* __restrict__ bf,
                                               float* __restrict__ feats) {
    int wave = threadIdx.x >> 5;
    int lane = threadIdx.x & 31;
    int tile = blockIdx.x * 8 + wave;    // 0..4095
    int tm   = tile >> 4;                // M tile: 0..255
    int tn   = tile & 15;                // N tile: 0..15
    int grp  = lane >> 4;                // half-wave
    int mn   = lane & 15;

    const float* arow = h  + ((size_t)(tm * 16 + mn)) * 64;  // A row m
    const float* brow = wf + ((size_t)(tn * 16 + mn)) * 64;  // B col n (wf is (256,64))

    v8f acc = {};
    #pragma unroll
    for (int kb = 0; kb < 64; kb += 32) {
        v16h a, bfrag;
        #pragma unroll
        for (int j = 0; j < 16; ++j) {
            // ISA 16-bit A 16x32 layout: K = (j&7) + 8*half-wave + 16*(j>=8)
            int k = (j & 7) + ((j >> 3) << 4) + grp * 8 + kb;
            a[j]     = (_Float16)arow[k];
            bfrag[j] = (_Float16)brow[k];
        }
        acc = __builtin_amdgcn_wmma_f32_16x16x32_f16(false, a, false, bfrag,
                                                     (short)0, acc, false, false);
    }
    #pragma unroll
    for (int v = 0; v < 8; ++v) {
        int m = tm * 16 + v + grp * 8;   // C/D layout: VGPR v -> M = v + 8*half
        int n = tn * 16 + mn;
        feats[(size_t)m * 256 + n] = fmaxf(acc[v] + bf[n], 0.f);
    }
}

// ---------------------------------------------------------------------------
// Per image: normalize feats -> 8-qubit circuit in LDS -> <Z0> -> scalar MLP
// -> sigmoid.  One 128-thread block per image; state = 256 floats in LDS.
// ---------------------------------------------------------------------------
__global__ __launch_bounds__(128) void quantum_mlp(const float* __restrict__ feats,
        const float* __restrict__ qw,
        const float* __restrict__ wc1, const float* __restrict__ bc1,
        const float* __restrict__ wc2, const float* __restrict__ bc2,
        const float* __restrict__ wc3, const float* __restrict__ bc3,
        const float* __restrict__ wc4, const float* __restrict__ bc4,
        const float* __restrict__ wc5, const float* __restrict__ bc5,
        float* __restrict__ out) {
    __shared__ float st[256];
    __shared__ float red[128];
    __shared__ float m1[200], m2[152], m3[104], m4[56];
    int t = threadIdx.x, b = blockIdx.x;

    float f0 = feats[(size_t)b * 256 + t];
    float f1 = feats[(size_t)b * 256 + 128 + t];
    red[t] = f0 * f0 + f1 * f1;
    __syncthreads();
    for (int s = 64; s > 0; s >>= 1) { if (t < s) red[t] += red[t + s]; __syncthreads(); }
    float inv = 1.f / fmaxf(sqrtf(red[0]), 1e-12f);   // double-normalize == single
    st[t] = f0 * inv; st[t + 128] = f1 * inv;

    for (int l = 0; l < 7; ++l) {
        for (int q = 0; q < 8; ++q) {                 // RY(theta) on qubit q (MSB-first)
            __syncthreads();
            float th = qw[l * 8 + q] * 0.5f;
            float cth = cosf(th), sth = sinf(th);
            unsigned bit = 1u << (7 - q);
            unsigned low = (unsigned)t & (bit - 1);
            unsigned i0  = (((unsigned)t & ~(bit - 1)) << 1) | low;
            unsigned i1  = i0 | bit;
            float a0 = st[i0], a1 = st[i1];
            st[i0] = cth * a0 - sth * a1;
            st[i1] = sth * a0 + cth * a1;
        }
        for (int q = 0; q < 7; ++q) {                 // CNOT(q, q+1): swap where ctrl=1
            __syncthreads();
            if (t < 64) {
                int pt = 6 - q;                       // target bit position
                unsigned low  = (unsigned)t & ((1u << pt) - 1);
                unsigned high = ((unsigned)t) >> pt;
                unsigned i0 = (high << (pt + 2)) | (1u << (pt + 1)) | low; // ctrl=1,tgt=0
                unsigned i1 = i0 | (1u << pt);
                float a = st[i0]; st[i0] = st[i1]; st[i1] = a;
            }
        }
    }
    __syncthreads();
    red[t] = st[t] * st[t] - st[t + 128] * st[t + 128];   // <Z_0>
    __syncthreads();
    for (int s = 64; s > 0; s >>= 1) { if (t < s) red[t] += red[t + s]; __syncthreads(); }
    float qv = red[0];

    for (int j = t; j < 200; j += 128) m1[j] = fmaxf(qv * wc1[j] + bc1[j], 0.f);
    __syncthreads();
    for (int j = t; j < 150; j += 128) {
        float a = bc2[j];
        for (int k = 0; k < 200; ++k) a = fmaf(wc2[j * 200 + k], m1[k], a);
        m2[j] = fmaxf(a, 0.f);
    }
    __syncthreads();
    for (int j = t; j < 100; j += 128) {
        float a = bc3[j];
        for (int k = 0; k < 150; ++k) a = fmaf(wc3[j * 150 + k], m2[k], a);
        m3[j] = fmaxf(a, 0.f);
    }
    __syncthreads();
    if (t < 50) {
        float a = bc4[t];
        for (int k = 0; k < 100; ++k) a = fmaf(wc4[t * 100 + k], m3[k], a);
        m4[t] = fmaxf(a, 0.f);
    }
    __syncthreads();
    if (t == 0) {
        float a = bc5[0];
        for (int k = 0; k < 50; ++k) a = fmaf(wc5[k], m4[k], a);
        out[b] = 1.f / (1.f + expf(-a));
    }
}

// ---------------------------------------------------------------------------
extern "C" void kernel_launch(void* const* d_in, const int* in_sizes, int n_in,
                              void* d_out, int out_size, void* d_ws, size_t ws_size,
                              hipStream_t stream) {
    const float* x   = (const float*)d_in[0];
    const float* w1  = (const float*)d_in[1];
    const float* b1  = (const float*)d_in[2];
    const float* w2  = (const float*)d_in[3];
    const float* b2  = (const float*)d_in[4];
    const float* w3  = (const float*)d_in[5];
    const float* b3  = (const float*)d_in[6];
    const float* wf  = (const float*)d_in[7];
    const float* bf  = (const float*)d_in[8];
    const float* qw  = (const float*)d_in[9];
    const float* wc1 = (const float*)d_in[10];
    const float* bc1 = (const float*)d_in[11];
    const float* wc2 = (const float*)d_in[12];
    const float* bc2 = (const float*)d_in[13];
    const float* wc3 = (const float*)d_in[14];
    const float* bc3 = (const float*)d_in[15];
    const float* wc4 = (const float*)d_in[16];
    const float* bc4 = (const float*)d_in[17];
    const float* wc5 = (const float*)d_in[18];
    const float* bc5 = (const float*)d_in[19];
    float* outp = (float*)d_out;
    (void)in_sizes; (void)n_in; (void)out_size; (void)ws_size;

    char* ws = (char*)d_ws;
    float* y1    = (float*)(ws);                                  // 16.78M f
    float* y2    = (float*)(ws + (size_t)16777216 * 4);           // 8.39M f
    float* hbuf  = (float*)(ws);                                  // reuse (y1 dead)
    float* feats = (float*)(ws + (size_t)4 * 1024 * 1024);        // 1.05M f

    conv1_relu <<<65536, 256, 0, stream>>>(x,  w1, b1, y1);
    conv2_relu <<<32768, 256, 0, stream>>>(y1, w2, b2, y2);
    conv3_pool <<< 4096, 256, 0, stream>>>(y2, w3, b3, hbuf);
    fc_wmma    <<<  512, 256, 0, stream>>>(hbuf, wf, bf, feats);
    quantum_mlp<<< 4096, 128, 0, stream>>>(feats, qw, wc1, bc1, wc2, bc2,
                                           wc3, bc3, wc4, bc4, wc5, bc5, outp);
}
